// RotaryMultiheadSelfAttention_47785806135605
// MI455X (gfx1250) — matrix operand order repaired and checked by reading, not compiled, not run on hardware
//
#include <hip/hip_runtime.h>

typedef __bf16 bf16_t;
typedef __attribute__((ext_vector_type(8)))  __bf16 v8bf;
typedef __attribute__((ext_vector_type(16))) __bf16 v16bf;
typedef __attribute__((ext_vector_type(8)))  float  v8f;

#define SEQ      2048
#define BATCH    2
#define DMODEL   1024
#define NHEAD    16
#define HDIM     64
#define MTOT     (BATCH * SEQ)   // 4096

// ---------------------------------------------------------------------------
// CDNA5 async global->LDS copy (ASYNCcnt-tracked), guarded so we always have
// a working fallback if the toolchain lacks the builtin.
// Probe-learned signature: (int4 AS1* gsrc, int4 AS3* ldsdst, Ii offset, Ii cpol)
// ---------------------------------------------------------------------------
#if __has_builtin(__builtin_amdgcn_global_load_async_to_lds_b128)
#define USE_ASYNC_LDS 1
typedef int v4i_gcc __attribute__((vector_size(16)));
typedef __attribute__((address_space(1))) v4i_gcc as1_v4i;
typedef __attribute__((address_space(3))) v4i_gcc as3_v4i;
static __device__ __forceinline__ void async_cp16(bf16_t* l, const bf16_t* g) {
  __builtin_amdgcn_global_load_async_to_lds_b128(
      (as1_v4i*)const_cast<bf16_t*>(g), (as3_v4i*)l, 0, 0);
}
static __device__ __forceinline__ void async_wait0() {
#if __has_builtin(__builtin_amdgcn_s_wait_asynccnt)
  __builtin_amdgcn_s_wait_asynccnt(0);
#else
  asm volatile("s_wait_asynccnt 0x0" ::: "memory");
#endif
}
#else
#define USE_ASYNC_LDS 0
#endif

// ---------------------------------------------------------------------------
// WMMA helpers (CDNA5 gfx1250, wave32).
// A-fragment (16x32 bf16): lane holds row M=lane%16; K chunks at +hb8, +16+hb8,
//   hb8=(lane>=16)?8:0.
// B-fragment (32x16 bf16): lane holds col N=lane%16; K chunks at +hb8, +16+hb8,
//   hb8=(lane>=16)?8:0 (same order as A).
// C/D (16x16 f32): element r -> (M = r + (lane>=16?8:0), N = lane%16).
// ---------------------------------------------------------------------------
static __device__ __forceinline__ v16bf concat8(v8bf lo, v8bf hi) {
  return __builtin_shufflevector(lo, hi, 0, 1, 2, 3, 4, 5, 6, 7,
                                 8, 9, 10, 11, 12, 13, 14, 15);
}

static __device__ __forceinline__ v16bf load_a_frag(const bf16_t* p, int stride,
                                                    int row, int hb8) {
  const bf16_t* q = p + (size_t)row * stride + hb8;
  v8bf lo = *(const v8bf*)q;
  v8bf hi = *(const v8bf*)(q + 16);
  return concat8(lo, hi);
}

static __device__ __forceinline__ v16bf load_b_frag(const bf16_t* p, int stride,
                                                    int col, int hb16) {
  const bf16_t* q = p + (size_t)col * stride + (hb16 >> 1);
  v8bf lo = *(const v8bf*)q;
  v8bf hi = *(const v8bf*)(q + 16);
  return concat8(lo, hi);
}

static __device__ __forceinline__ v8f wmma_bf16(v16bf a, v16bf b, v8f c) {
  return __builtin_amdgcn_wmma_f32_16x16x32_bf16(false, a, false, b,
                                                 (short)0, c, false, false);
}

// ---------------------------------------------------------------------------
// Stage 0: fp32 -> bf16 converts (plain + transpose for weights)
// ---------------------------------------------------------------------------
__global__ void cvt_f32_bf16_kernel(const float* __restrict__ src,
                                    bf16_t* __restrict__ dst, int n) {
  int i = blockIdx.x * 256 + threadIdx.x;
  if (i < n) dst[i] = (bf16_t)src[i];
}

// dst[n*K + k] = src[k*N + n]   (weights stored [K,N] -> bf16 [N,K])
__global__ void transpose_f32_bf16_kernel(const float* __restrict__ src,
                                          bf16_t* __restrict__ dst,
                                          int K, int N) {
  int i = blockIdx.x * 256 + threadIdx.x;
  if (i < K * N) {
    int n = i / K;
    int k = i - n * K;
    dst[i] = (bf16_t)src[(size_t)k * N + n];
  }
}

// ---------------------------------------------------------------------------
// Shared GEMM tile loop body: 256 thr / 8 waves, tile 128x128, K-step 32.
// Stages A (rows of X) and B (rows of W^T) 128x32 bf16 tiles into LDS.
// ---------------------------------------------------------------------------
#define GEMM_PROLOG()                                                        \
  __shared__ bf16_t As[128 * 32];                                            \
  __shared__ bf16_t Bs[128 * 32];                                            \
  const int tid = threadIdx.x;                                               \
  const int wave = tid >> 5, lane = tid & 31;                                \
  const int ln = lane & 15;                                                  \
  const int hb8 = (lane >> 4) << 3;                                          \
  const int hb16 = (lane >> 4) << 4;                                         \
  const int waveM = wave & 3;                                                \
  const int waveN = wave >> 2;                                               \
  const int m0 = blockIdx.y * 128;                                           \
  const int n0 = blockIdx.x * 128;                                           \
  v8f acc[2][4];                                                             \
  _Pragma("unroll") for (int i = 0; i < 2; ++i)                              \
      _Pragma("unroll") for (int j = 0; j < 4; ++j)                          \
      _Pragma("unroll") for (int r = 0; r < 8; ++r) acc[i][j][r] = 0.0f;     \
  const int lrow = tid >> 1;                                                 \
  const int lcol = (tid & 1) << 4;

#if USE_ASYNC_LDS
#define GEMM_STAGE(AP, BP)                                                   \
  __syncthreads();                                                           \
  async_cp16(As + lrow * 32 + lcol, (AP) + (size_t)(m0 + lrow) * DMODEL + k0 + lcol);      \
  async_cp16(As + lrow * 32 + lcol + 8, (AP) + (size_t)(m0 + lrow) * DMODEL + k0 + lcol + 8); \
  async_cp16(Bs + lrow * 32 + lcol, (BP) + (size_t)(n0 + lrow) * DMODEL + k0 + lcol);      \
  async_cp16(Bs + lrow * 32 + lcol + 8, (BP) + (size_t)(n0 + lrow) * DMODEL + k0 + lcol + 8); \
  if (k0 + 32 < DMODEL) {                                                    \
    __builtin_prefetch((AP) + (size_t)(m0 + lrow) * DMODEL + k0 + 32 + lcol, 0, 1); \
    __builtin_prefetch((BP) + (size_t)(n0 + lrow) * DMODEL + k0 + 32 + lcol, 0, 1); \
  }                                                                          \
  async_wait0();                                                             \
  __syncthreads();
#else
#define GEMM_STAGE(AP, BP)                                                   \
  v8bf a0 = *(const v8bf*)((AP) + (size_t)(m0 + lrow) * DMODEL + k0 + lcol); \
  v8bf a1 = *(const v8bf*)((AP) + (size_t)(m0 + lrow) * DMODEL + k0 + lcol + 8); \
  v8bf b0 = *(const v8bf*)((BP) + (size_t)(n0 + lrow) * DMODEL + k0 + lcol); \
  v8bf b1 = *(const v8bf*)((BP) + (size_t)(n0 + lrow) * DMODEL + k0 + lcol + 8); \
  if (k0 + 32 < DMODEL) {                                                    \
    __builtin_prefetch((AP) + (size_t)(m0 + lrow) * DMODEL + k0 + 32 + lcol, 0, 1); \
    __builtin_prefetch((BP) + (size_t)(n0 + lrow) * DMODEL + k0 + 32 + lcol, 0, 1); \
  }                                                                          \
  __syncthreads();                                                           \
  *(v8bf*)(As + lrow * 32 + lcol) = a0;                                      \
  *(v8bf*)(As + lrow * 32 + lcol + 8) = a1;                                  \
  *(v8bf*)(Bs + lrow * 32 + lcol) = b0;                                      \
  *(v8bf*)(Bs + lrow * 32 + lcol + 8) = b1;                                  \
  __syncthreads();
#endif

#define GEMM_COMPUTE()                                                       \
  v16bf af[2], bf_[4];                                                       \
  _Pragma("unroll") for (int fm = 0; fm < 2; ++fm)                           \
      af[fm] = load_a_frag(As, 32, waveM * 32 + fm * 16 + ln, hb8);          \
  _Pragma("unroll") for (int fn = 0; fn < 4; ++fn)                           \
      bf_[fn] = load_b_frag(Bs, 32, waveN * 64 + fn * 16 + ln, hb16);        \
  _Pragma("unroll") for (int fm = 0; fm < 2; ++fm)                           \
      _Pragma("unroll") for (int fn = 0; fn < 4; ++fn)                       \
          acc[fm][fn] = wmma_bf16(af[fm], bf_[fn], acc[fm][fn]);

// ---------------------------------------------------------------------------
// Stage 1: QKV = X @ Wqkv + b, fused RoPE, scatter to per-head layouts.
// ---------------------------------------------------------------------------
__global__ void __launch_bounds__(256)
qkv_gemm_rope_kernel(const bf16_t* __restrict__ X,   // [4096,1024]
                     const bf16_t* __restrict__ WT,  // [3072,1024] (W^T)
                     const float* __restrict__ bias, // [3072]
                     bf16_t* __restrict__ Qp,        // [B,H,S,64]
                     bf16_t* __restrict__ Kp,        // [B,H,S,64]
                     bf16_t* __restrict__ Vt) {      // [B,H,64,S]
  GEMM_PROLOG();

  for (int k0 = 0; k0 < DMODEL; k0 += 32) {
    GEMM_STAGE(X, WT);
    GEMM_COMPUTE();
  }

  // Epilogue: bias + RoPE (Q,K only) + scatter.
  const float ropeC = -0.28782313662425576f;  // -2*ln(10000)/64
#pragma unroll
  for (int fm = 0; fm < 2; ++fm) {
#pragma unroll
    for (int fn = 0; fn < 4; ++fn) {
      const int n = n0 + waveN * 64 + fn * 16 + ln;
      const float bv = bias[n];
      const int d = n & 63;
      const int sect = n >> 10;        // 0=Q, 1=K, 2=V (uniform per fragment)
      const int h = (n >> 6) & 15;     // uniform per fragment
      const float theta = __expf(ropeC * (float)(d >> 2));
#pragma unroll
      for (int r = 0; r < 8; ++r) {
        const int m = m0 + waveM * 32 + fm * 16 + r + hb8;
        const int b_ = m >> 11;
        const int s = m & (SEQ - 1);
        float v = acc[fm][fn][r] + bv;
        if (sect < 2) {
          const float vp = __shfl_xor(v, 1);
          const float ang = (float)s * theta;
          const float c = __cosf(ang), sn = __sinf(ang);
          v = (d & 1) ? (vp * sn + v * c) : (v * c - vp * sn);
        }
        if (sect == 0) {
          Qp[((size_t)((b_ * NHEAD + h) * SEQ + s)) * HDIM + d] = (bf16_t)v;
        } else if (sect == 1) {
          Kp[((size_t)((b_ * NHEAD + h) * SEQ + s)) * HDIM + d] = (bf16_t)v;
        } else {
          Vt[((size_t)((b_ * NHEAD + h) * HDIM + d)) * SEQ + s] = (bf16_t)v;
        }
      }
    }
  }
}

// ---------------------------------------------------------------------------
// Stage 2: flash attention. Block 128 thr = 4 waves, all sharing one (b,h).
// K/V tiles (32 keys) staged once per block in LDS; each wave owns 16 queries.
// ---------------------------------------------------------------------------
__global__ void __launch_bounds__(128)
attn_kernel(const bf16_t* __restrict__ Qp, const bf16_t* __restrict__ Kp,
            const bf16_t* __restrict__ Vt, bf16_t* __restrict__ Hbf) {
  __shared__ bf16_t Ks[32 * 64];      // [key][d]
  __shared__ bf16_t Vs[64 * 32];      // [d][key]
  __shared__ bf16_t Pl[4][16 * 32];   // per-wave P staging

  const int tid = threadIdx.x;
  const int wave = tid >> 5, lane = tid & 31;
  const int ln = lane & 15;
  const int hb8 = (lane >> 4) << 3;
  const int hb16 = (lane >> 4) << 4;
  const int bh = blockIdx.y;
  const int q0 = blockIdx.x * 64 + wave * 16;

  const bf16_t* Qb = Qp + (size_t)bh * SEQ * HDIM;
  const bf16_t* Kb = Kp + (size_t)bh * SEQ * HDIM;
  const bf16_t* Vb = Vt + (size_t)bh * HDIM * SEQ;
  bf16_t* P = Pl[wave];

  // staging coordinates: K tile = 32 rows x 64 d; V tile = 64 rows(d) x 32 keys
  const int krow = tid >> 2, kq = (tid & 3) << 4;   // 32B per thread
  const int drow = tid >> 1, dq = (tid & 1) << 4;

  v16bf qa[2];
#pragma unroll
  for (int f = 0; f < 2; ++f)
    qa[f] = load_a_frag(Qb + (size_t)q0 * HDIM + f * 32, HDIM, ln, hb8);

  v8f o[4];
#pragma unroll
  for (int f = 0; f < 4; ++f)
#pragma unroll
    for (int r = 0; r < 8; ++r) o[f][r] = 0.0f;
  float mrow[8], lrow_[8];
#pragma unroll
  for (int r = 0; r < 8; ++r) { mrow[r] = -3.0e38f; lrow_[r] = 0.0f; }

  for (int kb = 0; kb < SEQ; kb += 32) {
#if USE_ASYNC_LDS
    __syncthreads();
    async_cp16(Ks + krow * 64 + kq, Kb + (size_t)(kb + krow) * HDIM + kq);
    async_cp16(Ks + krow * 64 + kq + 8, Kb + (size_t)(kb + krow) * HDIM + kq + 8);
    async_cp16(Vs + drow * 32 + dq, Vb + (size_t)drow * SEQ + kb + dq);
    async_cp16(Vs + drow * 32 + dq + 8, Vb + (size_t)drow * SEQ + kb + dq + 8);
    if (kb + 32 < SEQ) {
      __builtin_prefetch(Kb + (size_t)(kb + 32 + krow) * HDIM + kq, 0, 1);
      __builtin_prefetch(Vb + (size_t)drow * SEQ + kb + 32 + dq, 0, 1);
    }
    async_wait0();
    __syncthreads();
#else
    v8bf kr0 = *(const v8bf*)(Kb + (size_t)(kb + krow) * HDIM + kq);
    v8bf kr1 = *(const v8bf*)(Kb + (size_t)(kb + krow) * HDIM + kq + 8);
    v8bf vr0 = *(const v8bf*)(Vb + (size_t)drow * SEQ + kb + dq);
    v8bf vr1 = *(const v8bf*)(Vb + (size_t)drow * SEQ + kb + dq + 8);
    if (kb + 32 < SEQ) {
      __builtin_prefetch(Kb + (size_t)(kb + 32 + krow) * HDIM + kq, 0, 1);
      __builtin_prefetch(Vb + (size_t)drow * SEQ + kb + 32 + dq, 0, 1);
    }
    __syncthreads();
    *(v8bf*)(Ks + krow * 64 + kq) = kr0;
    *(v8bf*)(Ks + krow * 64 + kq + 8) = kr1;
    *(v8bf*)(Vs + drow * 32 + dq) = vr0;
    *(v8bf*)(Vs + drow * 32 + dq + 8) = vr1;
    __syncthreads();
#endif

    v8f s0, s1;
#pragma unroll
    for (int r = 0; r < 8; ++r) { s0[r] = 0.0f; s1[r] = 0.0f; }
#pragma unroll
    for (int f = 0; f < 2; ++f) {
      v16bf kf0 = load_b_frag(Ks + f * 32, HDIM, ln, hb16);
      s0 = wmma_bf16(qa[f], kf0, s0);
      v16bf kf1 = load_b_frag(Ks + f * 32, HDIM, 16 + ln, hb16);
      s1 = wmma_bf16(qa[f], kf1, s1);
    }

    float alpha[8];
#pragma unroll
    for (int r = 0; r < 8; ++r) {
      const float v0 = s0[r] * 0.125f;
      const float v1 = s1[r] * 0.125f;
      float t = fmaxf(v0, v1);
      t = fmaxf(t, __shfl_xor(t, 1));
      t = fmaxf(t, __shfl_xor(t, 2));
      t = fmaxf(t, __shfl_xor(t, 4));
      t = fmaxf(t, __shfl_xor(t, 8));
      const float mn = fmaxf(mrow[r], t);
      const float al = __expf(mrow[r] - mn);
      const float p0 = __expf(v0 - mn);
      const float p1 = __expf(v1 - mn);
      float rs = p0 + p1;
      rs += __shfl_xor(rs, 1);
      rs += __shfl_xor(rs, 2);
      rs += __shfl_xor(rs, 4);
      rs += __shfl_xor(rs, 8);
      lrow_[r] = lrow_[r] * al + rs;
      mrow[r] = mn;
      alpha[r] = al;
      const int M = r + hb8;
      P[M * 32 + ln] = (bf16_t)p0;          // C-layout -> A-layout via LDS
      P[M * 32 + 16 + ln] = (bf16_t)p1;
    }
#pragma unroll
    for (int f = 0; f < 4; ++f)
#pragma unroll
      for (int r = 0; r < 8; ++r) o[f][r] *= alpha[r];

    // Same-wave LDS RAW: DS ops are in-order within a wave.
    v16bf pa = load_a_frag(P, 32, ln, hb8);
#pragma unroll
    for (int f = 0; f < 4; ++f) {
      v16bf vf = load_b_frag(Vs, 32, f * 16 + ln, hb16);
      o[f] = wmma_bf16(pa, vf, o[f]);
    }
  }

  const int b_ = bh >> 4, h = bh & 15;
#pragma unroll
  for (int f = 0; f < 4; ++f) {
#pragma unroll
    for (int r = 0; r < 8; ++r) {
      const float val = o[f][r] / lrow_[r];
      const int m = q0 + r + hb8;
      const int col = h * HDIM + f * 16 + ln;
      Hbf[(size_t)(b_ * SEQ + m) * DMODEL + col] = (bf16_t)val;
    }
  }
}

// ---------------------------------------------------------------------------
// Stage 3: out = H @ Wo + b_o   (fp32 output)
// ---------------------------------------------------------------------------
__global__ void __launch_bounds__(256)
oproj_gemm_kernel(const bf16_t* __restrict__ H,    // [4096,1024]
                  const bf16_t* __restrict__ WT,   // [1024,1024] (W_o^T)
                  const float* __restrict__ bias,  // [1024]
                  float* __restrict__ out) {       // [4096,1024]
  GEMM_PROLOG();

  for (int k0 = 0; k0 < DMODEL; k0 += 32) {
    GEMM_STAGE(H, WT);
    GEMM_COMPUTE();
  }

#pragma unroll
  for (int fm = 0; fm < 2; ++fm) {
#pragma unroll
    for (int fn = 0; fn < 4; ++fn) {
      const int n = n0 + waveN * 64 + fn * 16 + ln;
      const float bv = bias[n];
#pragma unroll
      for (int r = 0; r < 8; ++r) {
        const int m = m0 + waveM * 32 + fm * 16 + r + hb8;
        out[(size_t)m * DMODEL + n] = acc[fm][fn][r] + bv;
      }
    }
  }
}

// ---------------------------------------------------------------------------
// Launcher. Workspace layout (bytes):
//   [0,8M)    Xbf    bf16 [4096,1024]
//   [8M,14M)  WqkvT  bf16 [3072,1024]
//   [14M,16M) WoT    bf16 [1024,1024]
//   [16M,24M) Qp     bf16 [B,H,S,64]
//   [24M,32M) Kp     bf16 [B,H,S,64]
//   [32M,40M) Vt     bf16 [B,H,64,S]
//   [40M,48M) Hbf    bf16 [4096,1024]
// ---------------------------------------------------------------------------
extern "C" void kernel_launch(void* const* d_in, const int* in_sizes, int n_in,
                              void* d_out, int out_size, void* d_ws,
                              size_t ws_size, hipStream_t stream) {
  (void)in_sizes; (void)n_in; (void)out_size; (void)ws_size;
  const float* emb = (const float*)d_in[0];
  const float* W_qkv = (const float*)d_in[1];
  const float* b_qkv = (const float*)d_in[2];
  const float* W_o = (const float*)d_in[3];
  const float* b_o = (const float*)d_in[4];
  float* out = (float*)d_out;

  char* ws = (char*)d_ws;
  bf16_t* Xbf = (bf16_t*)(ws);
  bf16_t* WqkvT = (bf16_t*)(ws + (size_t)(8 << 20));
  bf16_t* WoT = (bf16_t*)(ws + (size_t)(14 << 20));
  bf16_t* Qp = (bf16_t*)(ws + (size_t)(16 << 20));
  bf16_t* Kp = (bf16_t*)(ws + (size_t)(24 << 20));
  bf16_t* Vt = (bf16_t*)(ws + (size_t)(32 << 20));
  bf16_t* Hbf = (bf16_t*)(ws + (size_t)(40 << 20));

  cvt_f32_bf16_kernel<<<(MTOT * DMODEL) / 256, 256, 0, stream>>>(
      emb, Xbf, MTOT * DMODEL);
  transpose_f32_bf16_kernel<<<(3 * DMODEL * DMODEL) / 256, 256, 0, stream>>>(
      W_qkv, WqkvT, DMODEL, 3 * DMODEL);
  transpose_f32_bf16_kernel<<<(DMODEL * DMODEL) / 256, 256, 0, stream>>>(
      W_o, WoT, DMODEL, DMODEL);

  qkv_gemm_rope_kernel<<<dim3(24, 32), 256, 0, stream>>>(Xbf, WqkvT, b_qkv, Qp,
                                                         Kp, Vt);
  attn_kernel<<<dim3(SEQ / 64, BATCH * NHEAD), 128, 0, stream>>>(Qp, Kp, Vt,
                                                                 Hbf);
  oproj_gemm_kernel<<<dim3(8, 32), 256, 0, stream>>>(Hbf, WoT, b_o, out);
}
